// AddRNN_68418829025354
// MI455X (gfx1250) — matrix operand order repaired and checked by reading, not compiled, not run on hardware
//
#include <hip/hip_runtime.h>
#include <math.h>

#define EPSF 1e-5f
#define MAXN 0.999f   // (1 - 1e-3) / sqrt(C), C = 1

typedef __attribute__((ext_vector_type(16))) __bf16       bf16x16;
typedef __attribute__((ext_vector_type(8)))  float        f32x8;
typedef __attribute__((ext_vector_type(4)))  unsigned int u32x4;

union Frag { u32x4 u[2]; bf16x16 v; };

__device__ __forceinline__ unsigned short f2bf(float f) {
  unsigned int u = __float_as_uint(f);
  unsigned int r = (u + 0x7FFFu + ((u >> 16) & 1u)) >> 16;  // RNE
  return (unsigned short)r;
}
__device__ __forceinline__ float bf2f(unsigned short h) {
  return __uint_as_float(((unsigned int)h) << 16);
}
__device__ __forceinline__ float artanh_c(float x) {
  x = fminf(fmaxf(x, -1.0f + EPSF), 1.0f - EPSF);
  return 0.5f * logf((1.0f + x) / (1.0f - x));
}

// -------- block-wide sum over 256 threads --------
__device__ float block_sum_256(float v) {
  __shared__ float sd[256];
  int t = threadIdx.x;
  sd[t] = v;
  __syncthreads();
  #pragma unroll
  for (int s = 128; s > 0; s >>= 1) {
    if (t < s) sd[t] += sd[t + s];
    __syncthreads();
  }
  float r = sd[0];
  __syncthreads();
  return r;
}

// -------- Mobius add (in place on x) over 1024-length row, 4 elems/thread ----
__device__ void madd4(float* x, const float* y) {
  float sx = 0.f, sy = 0.f, sxy = 0.f;
  #pragma unroll
  for (int i = 0; i < 4; ++i) { sx += x[i]*x[i]; sy += y[i]*y[i]; sxy += x[i]*y[i]; }
  float x2 = block_sum_256(sx);
  float y2 = block_sum_256(sy);
  float xy = block_sum_256(sxy);
  float cx  = 1.f + 2.f*xy + y2;
  float cy  = 1.f - x2;
  float den = fmaxf(1.f + 2.f*xy + x2*y2, 1e-12f);
  float inv = 1.f / den;
  float sn = 0.f;
  #pragma unroll
  for (int i = 0; i < 4; ++i) { x[i] = (cx*x[i] + cy*y[i]) * inv; sn += x[i]*x[i]; }
  float n2 = block_sum_256(sn);
  float n  = sqrtf(fmaxf(n2, 1e-12f));
  float s  = fminf(1.f, MAXN / n);          // _project
  #pragma unroll
  for (int i = 0; i < 4; ++i) x[i] *= s;
}

// -------- Mobius pointwise tanh: exp0(tanh(log0(x))) ----
__device__ void mtanh4(float* x) {
  float sn = 0.f;
  #pragma unroll
  for (int i = 0; i < 4; ++i) sn += x[i]*x[i];
  float n2 = block_sum_256(sn);
  float n  = sqrtf(fmaxf(n2, 1e-12f));
  float g  = artanh_c(n) / n;               // log0 scale
  float sv = 0.f;
  #pragma unroll
  for (int i = 0; i < 4; ++i) { x[i] = tanhf(g * x[i]); sv += x[i]*x[i]; }
  float nv2 = block_sum_256(sv);
  float nv  = sqrtf(fmaxf(nv2, 1e-12f));
  float tn  = tanhf(nv);
  float e   = tn / nv * fminf(1.f, MAXN / fmaxf(tn, 1e-6f)); // exp0 + project
  #pragma unroll
  for (int i = 0; i < 4; ++i) x[i] *= e;
}

// ============ WMMA GEMM: C[M,N] f32 = A_bf16[M,Kpad] @ Bsw (fragment-swizzled)
// One wave per 16x64 output strip: A fragment loaded once per K-step and fed
// to 4 WMMAs against 4 adjacent (contiguous 1KB) B fragment blocks.
// Dual-B: strips with tm < mSplitTiles use B0 (premise), else B1 (hypothesis).
__global__ __launch_bounds__(256)
void gemm_bf16_wmma(const unsigned short* __restrict__ A,
                    const unsigned short* __restrict__ B0,
                    const unsigned short* __restrict__ B1,
                    float* __restrict__ C,
                    int M, int N, int Kpad, int ldA,
                    int mSplitTiles, int totalTiles)
{
  int wave = threadIdx.x >> 5;
  int lane = threadIdx.x & 31;
  int tile = blockIdx.x * 8 + wave;
  if (tile >= totalTiles) return;
  int ntn4 = N >> 6;                        // 64-wide N strips
  int tm   = tile / ntn4;
  int tn4  = tile - tm * ntn4;
  const unsigned short* Bsw = (tm < mSplitTiles) ? B0 : B1;
  int ntn = N >> 4;                         // 16-wide blocks per K-row
  int hlf = lane >> 4;                      // half-wave selector
  int m   = (tm << 4) + (lane & 15);
  const unsigned short* arow = A + (size_t)m * ldA + hlf * 16;
  int nk = Kpad >> 5;
  f32x8 acc[4];
  #pragma unroll
  for (int j = 0; j < 4; ++j) acc[j] = (f32x8){0.f,0.f,0.f,0.f,0.f,0.f,0.f,0.f};
  for (int kt = 0; kt < nk; ++kt) {
    Frag a;
    const unsigned short* ap = arow + (kt << 5);
    a.u[0] = *(const u32x4*)(ap);           // VGPR0-3: K = kk + hlf*16 .. +7
    a.u[1] = *(const u32x4*)(ap + 8);       // VGPR4-7: K = kk+8+hlf*16 .. +7
    const unsigned short* bp =
        Bsw + (((size_t)kt * ntn + (tn4 << 2)) * 32 + lane) * 16;
    if (kt + 1 < nk) __builtin_prefetch(bp + (size_t)ntn * 512, 0, 1);
    Frag b[4];
    #pragma unroll
    for (int j = 0; j < 4; ++j) {
      b[j].u[0] = *(const u32x4*)(bp + j * 512);
      b[j].u[1] = *(const u32x4*)(bp + j * 512 + 8);
    }
    #pragma unroll
    for (int j = 0; j < 4; ++j) {
      acc[j] = __builtin_amdgcn_wmma_f32_16x16x32_bf16(
          false, a.v, false, b[j].v, (short)0, acc[j], false, false);
    }
  }
  int cn0 = (tn4 << 6) + (lane & 15);
  #pragma unroll
  for (int j = 0; j < 4; ++j) {
    #pragma unroll
    for (int r = 0; r < 8; ++r) {
      int cm = (tm << 4) + (hlf << 3) + r;  // C/D layout: VGPR r -> M = r + 8*half
      C[(size_t)cm * N + cn0 + j * 16] = acc[j][r];
    }
  }
}

// ============ Pre-swizzle f32 [K,N] weights into per-lane WMMA B fragments ====
// B 32x16 bf16 fragment: lane l holds column n=l&15, K = (l>>4)*16 + 2v,+1.
__global__ __launch_bounds__(256)
void swizzle_b(const float* __restrict__ W, unsigned short* __restrict__ out,
               int K, int Kpad, int N, int total)
{
  int idx = blockIdx.x * 256 + threadIdx.x;     // one ushort-pair per thread
  if (idx >= total) return;
  int v    = idx & 7;
  int lane = (idx >> 3) & 31;
  int blk  = idx >> 8;                          // kt*ntn + nt
  int ntn  = N >> 4;
  int kt = blk / ntn, nt = blk - kt * ntn;
  int n = (nt << 4) + (lane & 15);
  int k = (kt << 5) + ((lane >> 4) << 4) + (v << 1);
  float w0 = (k     < K) ? W[(size_t)k       * N + n] : 0.f;
  float w1 = (k + 1 < K) ? W[(size_t)(k + 1) * N + n] : 0.f;
  out[(size_t)idx * 2 + 0] = f2bf(w0);
  out[(size_t)idx * 2 + 1] = f2bf(w1);
}

// ============ Embedding gather + _project + pad to 320 + xnorm ===============
__global__ __launch_bounds__(256)
void gather_project(const int* __restrict__ toks, const float* __restrict__ emb,
                    unsigned short* __restrict__ X, float* __restrict__ xnorm)
{
  int row = blockIdx.x;                         // 0..B*S-1
  int tok = toks[row];
  const float* e = emb + (size_t)tok * 300;
  float vals[2]; float s = 0.f;
  #pragma unroll
  for (int i = 0; i < 2; ++i) {
    int j = threadIdx.x + i * 256;
    float val = (j < 300) ? e[j] : 0.f;
    vals[i] = val; s += val * val;
  }
  float n2 = block_sum_256(s);
  float n  = sqrtf(fmaxf(n2, 1e-12f));
  float sc = fminf(1.f, MAXN / n);
  #pragma unroll
  for (int i = 0; i < 2; ++i) {
    int j = threadIdx.x + i * 256;
    if (j < 320) X[(size_t)row * 320 + j] = f2bf(vals[i] * sc);
  }
  if (threadIdx.x == 0) xnorm[row] = fmaxf(n * sc, 1e-6f);
}

// ============ Mobius matvec epilogue for the precomputed x@U =================
__global__ __launch_bounds__(256)
void mobius_scale_rows(const float* __restrict__ U, const float* __restrict__ xnorm,
                       unsigned short* __restrict__ Mx)
{
  int row = blockIdx.x;
  const float* u = U + (size_t)row * 1024;
  float uv[4]; float s = 0.f;
  #pragma unroll
  for (int i = 0; i < 4; ++i) { uv[i] = u[threadIdx.x + i * 256]; s += uv[i]*uv[i]; }
  float un2 = block_sum_256(s);
  float un  = sqrtf(fmaxf(un2, 1e-12f));
  float xn  = xnorm[row];
  float tt  = tanhf(un / xn * artanh_c(xn));
  float f   = tt / un * fminf(1.f, MAXN / fmaxf(tt, 1e-6f));
  #pragma unroll
  for (int i = 0; i < 4; ++i)
    Mx[(size_t)row * 1024 + threadIdx.x + i * 256] = f2bf(uv[i] * f);
}

// ============ Fused RNN-step / hlinear combine ===============================
// rows 0..127 = premise RNN (batch b=row), rows 128..255 = hypothesis RNN.
__global__ __launch_bounds__(256)
void rnn_combine(const float* __restrict__ U2, const float* __restrict__ Hprev,
                 const unsigned short* __restrict__ Mxp, const unsigned short* __restrict__ Mxh,
                 const float* __restrict__ biasp, const float* __restrict__ biash,
                 const int* __restrict__ plen, const int* __restrict__ hlen,
                 float* __restrict__ H, unsigned short* __restrict__ Hbf,
                 float* __restrict__ Last, unsigned short* __restrict__ Lastbf,
                 int t, int has_mx, int do_tanh, int capture)
{
  int row = blockIdx.x;
  int b   = row & 127;
  int isH = row >> 7;
  int tid = threadIdx.x;
  const float* u  = U2    + (size_t)row * 1024;
  const float* hp = Hprev + (size_t)row * 1024;
  float cur[4], hv[4];
  float su = 0.f, sh = 0.f;
  #pragma unroll
  for (int i = 0; i < 4; ++i) {
    int j = tid + i * 256;
    cur[i] = u[j]; hv[i] = hp[j];
    su += cur[i]*cur[i]; sh += hv[i]*hv[i];
  }
  // Mobius matvec scaling of u = h @ W, with xn = |h|
  float un2 = block_sum_256(su);
  float xn2 = block_sum_256(sh);
  float un  = sqrtf(fmaxf(un2, 1e-12f));
  float xn  = sqrtf(fmaxf(xn2, 1e-12f));
  float tt  = tanhf(un / xn * artanh_c(xn));
  float f   = tt / un * fminf(1.f, MAXN / fmaxf(tt, 1e-6f));
  #pragma unroll
  for (int i = 0; i < 4; ++i) cur[i] *= f;

  if (has_mx) {                                  // madd with precomputed U(x_t)
    const unsigned short* mx = (isH ? Mxh : Mxp) + ((size_t)b * 256 + t) * 1024;
    float mv[4];
    #pragma unroll
    for (int i = 0; i < 4; ++i) mv[i] = bf2f(mx[tid + i * 256]);
    madd4(cur, mv);
  }
  {                                              // madd with hyperbolic bias
    const float* bias = isH ? biash : biasp;
    float bv[4];
    #pragma unroll
    for (int i = 0; i < 4; ++i) bv[i] = bias[tid + i * 256];
    madd4(cur, bv);
  }
  if (do_tanh) mtanh4(cur);

  #pragma unroll
  for (int i = 0; i < 4; ++i) {
    int j = tid + i * 256;
    H  [(size_t)row * 1024 + j] = cur[i];
    Hbf[(size_t)row * 1024 + j] = f2bf(cur[i]);
  }
  if (capture) {
    int len = isH ? hlen[b] : plen[b];
    if (t == len - 1) {
      #pragma unroll
      for (int i = 0; i < 4; ++i) {
        int j = tid + i * 256;
        Last  [(size_t)row * 1024 + j] = cur[i];
        Lastbf[(size_t)row * 1024 + j] = f2bf(cur[i]);
      }
    }
  }
}

// ============ rep = madd(p_rep, h_rep) =======================================
__global__ __launch_bounds__(256)
void madd_rows(const float* __restrict__ XY, float* __restrict__ out)
{
  int b = blockIdx.x;
  float x[4], y[4];
  #pragma unroll
  for (int i = 0; i < 4; ++i) {
    int j = threadIdx.x + i * 256;
    x[i] = XY[(size_t)b * 1024 + j];
    y[i] = XY[(size_t)(b + 128) * 1024 + j];
  }
  madd4(x, y);
  #pragma unroll
  for (int i = 0; i < 4; ++i)
    out[(size_t)b * 1024 + threadIdx.x + i * 256] = x[i];
}

// ============ Hyperbolic MLR logits ==========================================
__global__ __launch_bounds__(256)
void mlr_kernel(const float* __restrict__ rep, const float* __restrict__ Am,
                const float* __restrict__ Pm, float* __restrict__ out)
{
  int b = blockIdx.x;
  int tid = threadIdx.x;
  float xv[4];
  #pragma unroll
  for (int i = 0; i < 4; ++i) xv[i] = rep[(size_t)b * 1024 + tid + i * 256];
  for (int k = 0; k < 3; ++k) {
    float pv[4], av[4], z[4];
    float sp = 0.f, sa = 0.f;
    #pragma unroll
    for (int i = 0; i < 4; ++i) {
      int j = tid + i * 256;
      pv[i] = Pm[(size_t)k * 1024 + j];
      av[i] = Am[(size_t)k * 1024 + j];
      sp += pv[i]*pv[i]; sa += av[i]*av[i];
      z[i] = -pv[i];
    }
    float p2 = block_sum_256(sp);
    float a2 = block_sum_256(sa);
    madd4(z, xv);                                 // z = (-p) (+) x, projected
    float sz = 0.f, sza = 0.f;
    #pragma unroll
    for (int i = 0; i < 4; ++i) { sz += z[i]*z[i]; sza += z[i]*av[i]; }
    float z2 = block_sum_256(sz);
    float za = block_sum_256(sza);
    float an  = sqrtf(fmaxf(a2, 1e-12f));
    float lam = 2.f / fmaxf(1.f - p2, EPSF);
    float arg = 2.f * za / (fmaxf(1.f - z2, EPSF) * an);
    if (tid == 0) out[b * 3 + k] = lam * an * asinhf(arg);
  }
}

// ============ zero init of hidden state ======================================
__global__ void zero_state(float* __restrict__ h, unsigned short* __restrict__ hbf, int n)
{
  int i = blockIdx.x * 256 + threadIdx.x;
  if (i < n) { h[i] = 0.f; hbf[i] = 0; }
}

// =============================================================================
extern "C" void kernel_launch(void* const* d_in, const int* in_sizes, int n_in,
                              void* d_out, int out_size, void* d_ws, size_t ws_size,
                              hipStream_t stream)
{
  (void)in_sizes; (void)n_in; (void)out_size; (void)ws_size;
  const int*   premise = (const int*)  d_in[0];
  const int*   p_len   = (const int*)  d_in[1];
  const int*   hypoth  = (const int*)  d_in[2];
  const int*   h_len   = (const int*)  d_in[3];
  const float* emb     = (const float*)d_in[4];
  const float* Wp      = (const float*)d_in[5];
  const float* Up      = (const float*)d_in[6];
  const float* bp      = (const float*)d_in[7];
  const float* Wh      = (const float*)d_in[8];
  const float* Uh      = (const float*)d_in[9];
  const float* bh      = (const float*)d_in[10];
  const float* Wcp     = (const float*)d_in[11];
  const float* bcp     = (const float*)d_in[12];
  const float* Wch     = (const float*)d_in[13];
  const float* bch     = (const float*)d_in[14];
  const float* a_mlr   = (const float*)d_in[15];
  const float* p_mlr   = (const float*)d_in[16];

  const int Bb = 128, S = 256, E = 300, Ep = 320, H = 1024, BS = Bb * S;

  char* base = (char*)d_ws;
  size_t off = 0;
  auto take = [&](size_t bytes) -> void* {
    void* p = base + off;
    off = (off + bytes + 255) & ~(size_t)255;
    return p;
  };
  unsigned short* embX  = (unsigned short*)take((size_t)BS * Ep * 2);
  float*          u1    = (float*)         take((size_t)BS * H * 4);
  unsigned short* Mxp   = (unsigned short*)take((size_t)BS * H * 2);
  unsigned short* Mxh   = (unsigned short*)take((size_t)BS * H * 2);
  float*          xnrm  = (float*)         take((size_t)BS * 4);
  unsigned short* Uswp  = (unsigned short*)take((size_t)Ep * H * 2);
  unsigned short* Uswh  = (unsigned short*)take((size_t)Ep * H * 2);
  unsigned short* Wswp  = (unsigned short*)take((size_t)H * H * 2);
  unsigned short* Wswh  = (unsigned short*)take((size_t)H * H * 2);
  unsigned short* Wswcp = (unsigned short*)take((size_t)H * H * 2);
  unsigned short* Wswch = (unsigned short*)take((size_t)H * H * 2);
  float*          hst   = (float*)         take((size_t)256 * H * 4);
  unsigned short* hbf   = (unsigned short*)take((size_t)256 * H * 2);
  float*          u2    = (float*)         take((size_t)256 * H * 4);
  float*          last  = (float*)         take((size_t)256 * H * 4);
  unsigned short* lastbf= (unsigned short*)take((size_t)256 * H * 2);
  float*          rep2  = (float*)         take((size_t)256 * H * 4);
  float*          rep   = (float*)         take((size_t)128 * H * 4);

  // ---- one-time weight swizzles into WMMA B-fragment layout ----
  int totU = (Ep / 32) * (H / 16) * 256;     // 163840
  int totW = (H  / 32) * (H / 16) * 256;     // 524288
  swizzle_b<<<totU / 256, 256, 0, stream>>>(Up,  Uswp,  E, Ep, H, totU);
  swizzle_b<<<totU / 256, 256, 0, stream>>>(Uh,  Uswh,  E, Ep, H, totU);
  swizzle_b<<<totW / 256, 256, 0, stream>>>(Wp,  Wswp,  H, H,  H, totW);
  swizzle_b<<<totW / 256, 256, 0, stream>>>(Wh,  Wswh,  H, H,  H, totW);
  swizzle_b<<<totW / 256, 256, 0, stream>>>(Wcp, Wswcp, H, H,  H, totW);
  swizzle_b<<<totW / 256, 256, 0, stream>>>(Wch, Wswch, H, H,  H, totW);
  zero_state<<<(256 * H + 255) / 256, 256, 0, stream>>>(hst, hbf, 256 * H);

  // ---- hoisted input projections: Mx = mmatvec(x_t, U) for all t ----
  int tiles1 = (BS / 16) * (H / 64);         // 32768 strips -> 4096 WGs
  gather_project<<<BS, 256, 0, stream>>>(premise, emb, embX, xnrm);
  gemm_bf16_wmma<<<tiles1 / 8, 256, 0, stream>>>(embX, Uswp, Uswp, u1,
                                                 BS, H, Ep, Ep, 1 << 28, tiles1);
  mobius_scale_rows<<<BS, 256, 0, stream>>>(u1, xnrm, Mxp);

  gather_project<<<BS, 256, 0, stream>>>(hypoth, emb, embX, xnrm);
  gemm_bf16_wmma<<<tiles1 / 8, 256, 0, stream>>>(embX, Uswh, Uswh, u1,
                                                 BS, H, Ep, Ep, 1 << 28, tiles1);
  mobius_scale_rows<<<BS, 256, 0, stream>>>(u1, xnrm, Mxh);

  // ---- sequential recurrence: both RNNs stacked (rows 0-127 / 128-255) ----
  int tiles2 = (256 / 16) * (H / 64);        // 256 strips -> 32 WGs
  for (int t = 0; t < S; ++t) {
    gemm_bf16_wmma<<<tiles2 / 8, 256, 0, stream>>>(hbf, Wswp, Wswh, u2,
                                                   256, H, H, H, 8, tiles2);
    rnn_combine<<<256, 256, 0, stream>>>(u2, hst, Mxp, Mxh, bp, bh, p_len, h_len,
                                         hst, hbf, last, lastbf, t,
                                         /*has_mx=*/1, /*do_tanh=*/1, /*capture=*/1);
  }

  // ---- hyperbolic linear heads on last states ----
  gemm_bf16_wmma<<<tiles2 / 8, 256, 0, stream>>>(lastbf, Wswcp, Wswch, u2,
                                                 256, H, H, H, 8, tiles2);
  rnn_combine<<<256, 256, 0, stream>>>(u2, last, Mxp, Mxh, bcp, bch, p_len, h_len,
                                       rep2, hbf, last, lastbf, 0,
                                       /*has_mx=*/0, /*do_tanh=*/0, /*capture=*/0);

  // ---- rep = madd(p_rep, h_rep); logits = MLR(rep) ----
  madd_rows<<<128, 256, 0, stream>>>(rep2, rep);
  mlr_kernel<<<128, 256, 0, stream>>>(rep, a_mlr, p_mlr, (float*)d_out);
}
